// DropGIN_ogb_10101763080478
// MI455X (gfx1250) — matrix operand order repaired and checked by reading, not compiled, not run on hardware
//
#include <hip/hip_runtime.h>
#include <hip/hip_bf16.h>

// ---- problem constants (from reference) ----
#define NN     50000
#define EE     800000
#define F_INC  128
#define DIMC   64
#define OUTC   10
#define RUNSC  4
#define GGC    256
#define MR     (RUNSC * NN)      // 200000 rows for BN / GEMM
#define BN_EPS 1e-5f

typedef __attribute__((ext_vector_type(16))) __bf16 v16bf;
typedef __attribute__((ext_vector_type(8)))  float  v8f;

static inline int cdiv_l(long a, long b) { return (int)((a + b - 1) / b); }

// ---------------- elementwise helpers ----------------
__global__ void fill_zero_k(float* __restrict__ p, int n) {
    int i = blockIdx.x * blockDim.x + threadIdx.x;
    if (i < n) p[i] = 0.0f;
}

__global__ void copy4_k(float4* __restrict__ dst, const float4* __restrict__ src, int n4) {
    int i = blockIdx.x * blockDim.x + threadIdx.x;
    if (i < n4) dst[i] = src[i];
}

// h[r,n,:] = mask[r,n] ? 0 : x[n,:]   (F_INC = 128, vectorized by float4)
__global__ void drop_nodes_k(const float4* __restrict__ x4,
                             const unsigned char* __restrict__ mask,
                             float4* __restrict__ h4) {
    int i = blockIdx.x * blockDim.x + threadIdx.x;           // RUNS*N*32
    if (i >= RUNSC * NN * (F_INC / 4)) return;
    int fq = i & 31;                 // F_INC/4 = 32
    int rn = i >> 5;                 // r*N + n
    int n  = rn % NN;
    float4 v = make_float4(0.f, 0.f, 0.f, 0.f);
    if (!mask[rn]) v = x4[(size_t)n * 32 + fq];
    h4[i] = v;
}

// z[r,dst,f] += h[r,src,f]  for all runs; one thread per (edge, feature)
template <int LOGF>
__global__ void aggregate_edges_k(const float* __restrict__ h,
                                  float* __restrict__ z,
                                  const int* __restrict__ src,
                                  const int* __restrict__ dst) {
    const int F = 1 << LOGF;
    int i = blockIdx.x * blockDim.x + threadIdx.x;
    if (i >= (EE << LOGF)) return;
    int e = i >> LOGF;
    int f = i & (F - 1);
    int s = src[e], d = dst[e];
    size_t so = (size_t)s * F + f;
    size_t dofs = (size_t)d * F + f;
#pragma unroll
    for (int r = 0; r < RUNSC; ++r)
        __builtin_prefetch(&h[(size_t)r * NN * F + so], 0, 0);  // global_prefetch_b8
#pragma unroll
    for (int r = 0; r < RUNSC; ++r)
        atomicAdd(&z[(size_t)r * NN * F + dofs], h[(size_t)r * NN * F + so]);
}

__global__ void convert_w_k(const float* __restrict__ w, __bf16* __restrict__ o, int n) {
    int i = blockIdx.x * blockDim.x + threadIdx.x;
    if (i < n) o[i] = (__bf16)w[i];
}

// ---------------- WMMA GEMM: C[M,64] = A[M,K](f32->bf16) @ W[K,64](bf16) + bias ----------------
// block = 256 threads = 8 waves; block tile = 128 rows x 64 cols; wave = 16 rows x 64 cols.
template <int K>
__global__ __launch_bounds__(256)
void gemm_bf16_k(const float* __restrict__ A, const __bf16* __restrict__ Wg,
                 const float* __restrict__ bias, float* __restrict__ C, int Mrows) {
    __shared__ __align__(16) __bf16 Wl[K * 64];  // fragment-packed weight tile

    // Repack W row-major [K,64] into per-(k32-block, col-tile, lane) contiguous 16-elem
    // B fragments: lane = ((k32>>3)&1)*16 + (n&15); element j = (k32&7) + (k32>>4)*8 for column n.
    for (int idx = threadIdx.x; idx < K * 64; idx += 256) {
        int k = idx >> 6;
        int n = idx & 63;
        int kkblk = k >> 5;
        int k32   = k & 31;
        int lane  = (((k32 >> 3) & 1) << 4) | (n & 15);
        int j     = (k32 & 7) | ((k32 >> 4) << 3);
        int ct    = n >> 4;
        Wl[((((kkblk << 2) + ct) << 5) + lane) * 16 + j] = Wg[idx];
    }
    __syncthreads();

    const int wave = threadIdx.x >> 5;
    const int lane = threadIdx.x & 31;
    const int rbase = blockIdx.x * 128 + wave * 16;
    const int arow  = rbase + (lane & 15);
    const bool rowok = (arow < Mrows);
    const int kA = (lane & 16) ? 8 : 0;   // A layout: lanes 16-31 hold K 8..15 / 24..31

    v8f acc[4];
#pragma unroll
    for (int ct = 0; ct < 4; ++ct)
#pragma unroll
        for (int j = 0; j < 8; ++j) acc[ct][j] = 0.0f;

#pragma unroll
    for (int kk = 0; kk < K; kk += 32) {
        float4 a0 = make_float4(0, 0, 0, 0), a1 = a0, a2 = a0, a3 = a0;
        if (rowok) {
            const float4* ap = (const float4*)(A + (size_t)arow * K + kk + kA);
            a0 = ap[0]; a1 = ap[1];
            const float4* ap2 = (const float4*)(A + (size_t)arow * K + kk + kA + 16);
            a2 = ap2[0]; a3 = ap2[1];
        }
        v16bf af;
        af[0]  = (__bf16)a0.x; af[1]  = (__bf16)a0.y; af[2]  = (__bf16)a0.z; af[3]  = (__bf16)a0.w;
        af[4]  = (__bf16)a1.x; af[5]  = (__bf16)a1.y; af[6]  = (__bf16)a1.z; af[7]  = (__bf16)a1.w;
        af[8]  = (__bf16)a2.x; af[9]  = (__bf16)a2.y; af[10] = (__bf16)a2.z; af[11] = (__bf16)a2.w;
        af[12] = (__bf16)a3.x; af[13] = (__bf16)a3.y; af[14] = (__bf16)a3.z; af[15] = (__bf16)a3.w;

        const int kkblk = kk >> 5;
#pragma unroll
        for (int ct = 0; ct < 4; ++ct) {
            const uint4* bp =
                (const uint4*)&Wl[((((kkblk << 2) + ct) << 5) + lane) * 16];
            union { uint4 q[2]; v16bf v; } bu;
            bu.q[0] = bp[0];   // ds_load_b128
            bu.q[1] = bp[1];   // ds_load_b128
            acc[ct] = __builtin_amdgcn_wmma_f32_16x16x32_bf16(
                false, af, false, bu.v, (short)0, acc[ct], false, false);
        }
    }

    // D layout: lanes 0-15: VGPR j -> M=j ; lanes 16-31: VGPR j -> M=8+j ; N = lane&15
    const int rlo = rbase + ((lane >> 4) & 1) * 8;
    const int n0  = lane & 15;
#pragma unroll
    for (int ct = 0; ct < 4; ++ct) {
        int col = ct * 16 + n0;
        float bv = bias[col];
#pragma unroll
        for (int j = 0; j < 8; ++j) {
            int row = rlo + j;
            if (row < Mrows) C[(size_t)row * 64 + col] = acc[ct][j] + bv;
        }
    }
}

// ---------------- BatchNorm (training mode over all M rows) ----------------
__global__ __launch_bounds__(256)
void bn_stats_k(const float* __restrict__ X, float* __restrict__ stats, int Mrows) {
    __shared__ float ssum[256];
    __shared__ float ssq[256];
    int col = threadIdx.x & 63;
    int grp = threadIdx.x >> 6;      // 4 row groups
    int r0 = blockIdx.x * 2048;
    int rend = min(r0 + 2048, Mrows);
    float s = 0.f, q = 0.f;
    for (int r = r0 + grp; r < rend; r += 4) {
        float v = X[(size_t)r * 64 + col];
        s += v; q += v * v;
    }
    ssum[threadIdx.x] = s; ssq[threadIdx.x] = q;
    __syncthreads();
    if (grp == 0) {
        float ts = ssum[col] + ssum[64 + col] + ssum[128 + col] + ssum[192 + col];
        float tq = ssq[col]  + ssq[64 + col]  + ssq[128 + col]  + ssq[192 + col];
        atomicAdd(&stats[col], ts);
        atomicAdd(&stats[64 + col], tq);
    }
}

__global__ void bn_relu_k(float* __restrict__ X, const float* __restrict__ stats,
                          const float* __restrict__ g, const float* __restrict__ b, int total) {
    int i = blockIdx.x * blockDim.x + threadIdx.x;
    if (i >= total) return;
    int col = i & 63;
    const float invM = 1.0f / (float)MR;
    float m = stats[col] * invM;
    float v = stats[64 + col] * invM - m * m;
    float y = (X[i] - m) * rsqrtf(v + BN_EPS) * g[col] + b[col];
    X[i] = fmaxf(y, 0.0f);
}

// ---------------- pooling + FC heads ----------------
// pooled[batch[n], f] += mean_r h[r,n,f]
__global__ void pool_accum_k(const float* __restrict__ h, const int* __restrict__ batch,
                             float* __restrict__ pooled, int logF) {
    int i = blockIdx.x * blockDim.x + threadIdx.x;
    int F = 1 << logF;
    if (i >= (NN << logF)) return;
    int f = i & (F - 1);
    int n = i >> logF;
    float s = 0.f;
#pragma unroll
    for (int r = 0; r < RUNSC; ++r) s += h[((size_t)r * NN + n) * F + f];
    atomicAdd(&pooled[(size_t)batch[n] * F + f], s * (1.0f / RUNSC));
}

// out[g,o] += pooled[g,:] @ w[:,o] + b[o]
__global__ void fc_apply_k(const float* __restrict__ pooled, const float* __restrict__ w,
                           const float* __restrict__ b, float* __restrict__ out, int F) {
    int i = blockIdx.x * blockDim.x + threadIdx.x;
    if (i >= GGC * OUTC) return;
    int g = i / OUTC, o = i % OUTC;
    float s = b[o];
    for (int f = 0; f < F; ++f) s += pooled[(size_t)g * F + f] * w[f * OUTC + o];
    out[i] += s;
}

// ---------------- host orchestration ----------------
extern "C" void kernel_launch(void* const* d_in, const int* in_sizes, int n_in,
                              void* d_out, int out_size, void* d_ws, size_t ws_size,
                              hipStream_t stream) {
    (void)in_sizes; (void)n_in; (void)out_size; (void)ws_size;

    const float* x          = (const float*)d_in[0];
    const int*   edge       = (const int*)d_in[1];          // [2,E]: src then dst
    const int*   batch      = (const int*)d_in[2];
    const unsigned char* dm = (const unsigned char*)d_in[3];

    const float *w1[4], *b1[4], *g1[4], *be1[4], *w2[4], *b2[4], *go[4], *bo[4];
    int ii = 4;
    for (int c = 0; c < 4; ++c) {
        w1[c] = (const float*)d_in[ii++]; b1[c] = (const float*)d_in[ii++];
        g1[c] = (const float*)d_in[ii++]; be1[c] = (const float*)d_in[ii++];
        w2[c] = (const float*)d_in[ii++]; b2[c] = (const float*)d_in[ii++];
        go[c] = (const float*)d_in[ii++]; bo[c] = (const float*)d_in[ii++];
    }
    const float *fcw[5], *fcb[5];
    for (int f = 0; f < 5; ++f) {
        fcw[f] = (const float*)d_in[ii++]; fcb[f] = (const float*)d_in[ii++];
    }

    // workspace carve (256B aligned sections)
    char* wp = (char*)d_ws;
    auto take = [&](size_t bytes) -> char* {
        char* r = wp; wp += (bytes + 255) & ~(size_t)255; return r;
    };
    float*  bufH   = (float*)take((size_t)MR * F_INC * sizeof(float));
    float*  bufZ   = (float*)take((size_t)MR * F_INC * sizeof(float));
    float*  bufT   = (float*)take((size_t)MR * DIMC * sizeof(float));
    float*  pooled = (float*)take((size_t)GGC * F_INC * sizeof(float));
    float*  stats  = (float*)take(2 * DIMC * sizeof(float));
    __bf16* wbf    = (__bf16*)take((size_t)F_INC * DIMC * sizeof(__bf16));

    float* out = (float*)d_out;
    const int T = 256;

    fill_zero_k<<<cdiv_l(GGC * OUTC, T), T, 0, stream>>>(out, GGC * OUTC);

    // h0 = drop(x)
    drop_nodes_k<<<cdiv_l((long)RUNSC * NN * (F_INC / 4), T), T, 0, stream>>>(
        (const float4*)x, dm, (float4*)bufH);

    // head 0 (F = 128)
    fill_zero_k<<<cdiv_l(GGC * F_INC, T), T, 0, stream>>>(pooled, GGC * F_INC);
    pool_accum_k<<<cdiv_l((long)NN * F_INC, T), T, 0, stream>>>(bufH, batch, pooled, 7);
    fc_apply_k<<<cdiv_l(GGC * OUTC, T), T, 0, stream>>>(pooled, fcw[0], fcb[0], out, F_INC);

    float* pH = bufH;
    float* pZ = bufZ;
    for (int c = 0; c < 4; ++c) {
        const int F = (c == 0) ? F_INC : DIMC;

        // z = h + neighbor-sum
        copy4_k<<<cdiv_l((long)MR * F / 4, T), T, 0, stream>>>(
            (float4*)pZ, (const float4*)pH, MR * F / 4);
        if (F == F_INC)
            aggregate_edges_k<7><<<cdiv_l((long)EE << 7, T), T, 0, stream>>>(pH, pZ, edge, edge + EE);
        else
            aggregate_edges_k<6><<<cdiv_l((long)EE << 6, T), T, 0, stream>>>(pH, pZ, edge, edge + EE);

        // t = relu(BN(z @ w1 + b1))
        convert_w_k<<<cdiv_l(F * DIMC, T), T, 0, stream>>>(w1[c], wbf, F * DIMC);
        if (F == F_INC)
            gemm_bf16_k<128><<<cdiv_l(MR, 128), 256, 0, stream>>>(pZ, wbf, b1[c], bufT, MR);
        else
            gemm_bf16_k<64><<<cdiv_l(MR, 128), 256, 0, stream>>>(pZ, wbf, b1[c], bufT, MR);
        fill_zero_k<<<1, 128, 0, stream>>>(stats, 2 * DIMC);
        bn_stats_k<<<cdiv_l(MR, 2048), 256, 0, stream>>>(bufT, stats, MR);
        bn_relu_k<<<cdiv_l((long)MR * DIMC, T), T, 0, stream>>>(bufT, stats, g1[c], be1[c], MR * DIMC);

        // h' = relu(BN(t @ w2 + b2))  -> overwrite pZ (first M*64 floats)
        convert_w_k<<<cdiv_l(DIMC * DIMC, T), T, 0, stream>>>(w2[c], wbf, DIMC * DIMC);
        gemm_bf16_k<64><<<cdiv_l(MR, 128), 256, 0, stream>>>(bufT, wbf, b2[c], pZ, MR);
        fill_zero_k<<<1, 128, 0, stream>>>(stats, 2 * DIMC);
        bn_stats_k<<<cdiv_l(MR, 2048), 256, 0, stream>>>(pZ, stats, MR);
        bn_relu_k<<<cdiv_l((long)MR * DIMC, T), T, 0, stream>>>(pZ, stats, go[c], bo[c], MR * DIMC);

        // head c+1 (F = 64)
        fill_zero_k<<<cdiv_l(GGC * F_INC, T), T, 0, stream>>>(pooled, GGC * F_INC);
        pool_accum_k<<<cdiv_l((long)NN * DIMC, T), T, 0, stream>>>(pZ, batch, pooled, 6);
        fc_apply_k<<<cdiv_l(GGC * OUTC, T), T, 0, stream>>>(pooled, fcw[c + 1], fcb[c + 1], out, DIMC);

        // ping-pong: new h lives in pZ
        float* tmp = pH; pH = pZ; pZ = tmp;
    }
}
